// slam_model_85048942396038
// MI455X (gfx1250) — matrix operand order repaired and checked by reading, not compiled, not run on hardware
//
#include <hip/hip_runtime.h>
#include <hip/hip_bf16.h>
#include <stdint.h>

#define B_ 4
#define L_ 512
#define D_ 2048
#define V_ 32000
#define T_ 1024
#define M_ (B_*L_)                // 2048 flattened (b,l) rows
#define COLS_PER_WG 64
#define NCHUNK (V_/COLS_PER_WG)   // 500 workgroups over vocab
#define NPART (NCHUNK*4)          // 2000 (chunk, subtile) candidates per row

typedef __attribute__((ext_vector_type(16))) __bf16 v16bf;
typedef __attribute__((ext_vector_type(8)))  float  v8f;

union Frag { v16bf v; uint4 q[2]; };

__device__ __forceinline__ unsigned short f2bf(float f) {
    unsigned int u = __float_as_uint(f);
    u += 0x7FFFu + ((u >> 16) & 1u);          // round-to-nearest-even
    return (unsigned short)(u >> 16);
}

// ---------------------------------------------------------------------------
// Kernel 1: encoder_outs f32 -> bf16 staging copy (8.4 MB, L2-resident after)
// ---------------------------------------------------------------------------
__global__ void __launch_bounds__(256) cvt_a_bf16(const float* __restrict__ a,
                                                  unsigned short* __restrict__ ab) {
    int i = (blockIdx.x * 256 + threadIdx.x) * 4;
    float4 f = *(const float4*)(a + i);
    ushort4 o;
    o.x = f2bf(f.x); o.y = f2bf(f.y); o.z = f2bf(f.z); o.w = f2bf(f.w);
    *(ushort4*)(ab + i) = o;
}

// ---------------------------------------------------------------------------
// Kernel 2: fused (W chunk -> LDS bf16) + w_norm + GEMM(WMMA) + partial argmax
// grid = 500 WGs (64-col vocab chunk each), 256 threads = 8 wave32.
// Each wave: one 16-col subtile, FOUR M-tiles per B fragment (4 independent
// WMMAs per ds_load pair). A per-iteration memory clobber keeps B fragments
// re-loaded from LDS (prevents the cross-iteration hoist -> scratch spill).
// LDS: 64 x 2048 bf16 tile (256KB) + 64 inv-norm floats.
// ---------------------------------------------------------------------------
__global__ void __launch_bounds__(256, 1) sim_argmax(
    const unsigned short* __restrict__ a_bf16,   // [M_][D_]
    const float*          __restrict__ w,        // [V_][D_] f32
    float2*               __restrict__ partial)  // [M_][NPART] (val, idx-bits)
{
    extern __shared__ unsigned char smem_raw[];
    unsigned short* Wt   = (unsigned short*)smem_raw;                 // [64][D_]
    float*          invn = (float*)(smem_raw + COLS_PER_WG * D_ * 2); // [64]

    const int tid  = threadIdx.x;
    const int lane = tid & 31;
    const int wave = tid >> 5;                 // 0..7
    const int vbase = blockIdx.x * COLS_PER_WG;

    // ---- stage W chunk into LDS as bf16, fuse sum-of-squares (w_norm) ----
    {
        const int row  = tid >> 2;             // 0..63  (4 lanes per vocab row)
        const int part = tid & 3;
        const float* src = w + (size_t)(vbase + row) * D_ + part * 512;
        unsigned short* dst = Wt + row * D_ + part * 512;
        float ss = 0.f;
        #pragma unroll 4
        for (int k = 0; k < 512; k += 4) {
            float4 f = *(const float4*)(src + k);
            ss += f.x*f.x + f.y*f.y + f.z*f.z + f.w*f.w;
            ushort4 o; o.x=f2bf(f.x); o.y=f2bf(f.y); o.z=f2bf(f.z); o.w=f2bf(f.w);
            *(ushort4*)(dst + k) = o;
        }
        ss += __shfl_xor(ss, 1, 32);           // combine the 4 lanes of a row
        ss += __shfl_xor(ss, 2, 32);
        if (part == 0) invn[row] = rsqrtf(ss);
    }
    __syncthreads();

    const int sub      = wave & 3;             // 16-col subtile within chunk
    const int colInSub = lane & 15;
    const int half     = lane >> 4;
    const int phase    = wave >> 2;            // which M-tile quads this wave owns
    const float myInv  = invn[sub * 16 + colInSub];
    const int partIdx  = blockIdx.x * 4 + sub;

    // ISA 16-bit B layout: lanes 0-15 carry K kb..+7, kb+16..+23 of column `lane`,
    // lanes 16-31 carry K kb+8..+15, kb+24..+31 of column `lane-16`.
    const unsigned short* bCol = Wt + colInSub * D_ + half * 8;

    for (int j = 0; j < M_ / 128; ++j) {       // 16 iterations, 4 M-tiles each
        // Keep LDS B-fragment loads inside this iteration (no cross-iteration
        // CSE/hoist -> no scratch spills); live ranges stay local.
        asm volatile("" ::: "memory");

        const int mt0 = j * 8 + phase * 4;     // tiles mt0 .. mt0+3
        // ISA 16-bit A layout: lanes 0-15 row base+lane (K kb..+7, kb+16..+23),
        // lanes 16-31 same rows, K kb+8..+15, kb+24..+31.
        const unsigned short* aRow =
            a_bf16 + (size_t)(mt0 * 16 + colInSub) * D_ + half * 8;

        v8f acc[4] = {{}, {}, {}, {}};
        #pragma unroll 4
        for (int kb = 0; kb < D_; kb += 32) {
            Frag bf;
            bf.q[0] = *(const uint4*)(bCol + kb);
            bf.q[1] = *(const uint4*)(bCol + kb + 16);
            #pragma unroll
            for (int t = 0; t < 4; ++t) {
                Frag af;
                af.q[0] = *(const uint4*)(aRow + (size_t)t * 16 * D_ + kb);
                af.q[1] = *(const uint4*)(aRow + (size_t)t * 16 * D_ + kb + 16);
                acc[t] = __builtin_amdgcn_wmma_f32_16x16x32_bf16(
                    false, af.v, false, bf.v, (short)0, acc[t], false, false);
            }
        }

        // C layout: VGPR r = (row base+r, col lane) / (row base+8+r, col lane-16)
        #pragma unroll
        for (int t = 0; t < 4; ++t) {
            const int rowBase = (mt0 + t) * 16 + half * 8;
            #pragma unroll
            for (int r = 0; r < 8; ++r) {
                float v = acc[t][r] * myInv;   // sim * 1/||w||  (argmax-equivalent)
                float m = v;
                m = fmaxf(m, __shfl_xor(m, 1, 32));   // butterfly max within
                m = fmaxf(m, __shfl_xor(m, 2, 32));   // each 16-lane half
                m = fmaxf(m, __shfl_xor(m, 4, 32));
                m = fmaxf(m, __shfl_xor(m, 8, 32));
                unsigned int eq = (unsigned int)__ballot(v == m);
                if (colInSub == 0) {
                    unsigned int grp = half ? (eq >> 16) : (eq & 0xFFFFu);
                    int idx = vbase + sub * 16 + (__ffs(grp) - 1); // lowest-index tie
                    partial[(size_t)(rowBase + r) * NPART + partIdx] =
                        make_float2(m, __int_as_float(idx));
                }
            }
        }
    }
}

// ---------------------------------------------------------------------------
// Kernel 3: per-row reduce over 2000 candidates -> quantize_index, then
// gather embed_weight[idx] into blended[b, l, :]   (one block per row)
// ---------------------------------------------------------------------------
__global__ void __launch_bounds__(256) reduce_quant(
    const float2* __restrict__ partial, const float* __restrict__ w,
    float* __restrict__ out)
{
    __shared__ float sval[256];
    __shared__ int   sidx[256];
    const int row = blockIdx.x;
    const int tid = threadIdx.x;
    float best = -3.4e38f; int bidx = 0x7FFFFFFF;
    for (int c = tid; c < NPART; c += 256) {
        float2 p = partial[(size_t)row * NPART + c];
        int i = __float_as_int(p.y);
        bool take = (p.x > best) || (p.x == best && i < bidx);
        best = take ? p.x : best;
        bidx = take ? i   : bidx;
    }
    sval[tid] = best; sidx[tid] = bidx;
    __syncthreads();
    for (int s = 128; s > 0; s >>= 1) {
        if (tid < s) {
            float ov = sval[tid + s]; int oi = sidx[tid + s];
            bool take = (ov > sval[tid]) || (ov == sval[tid] && oi < sidx[tid]);
            sval[tid] = take ? ov : sval[tid];
            sidx[tid] = take ? oi : sidx[tid];
        }
        __syncthreads();
    }
    const int idx = sidx[0];
    const int b = row / L_, l = row % L_;
    if (tid == 0) out[(size_t)B_ * T_ * D_ + row] = (float)idx;   // index output
    const float* src = w + (size_t)idx * D_;
    float* dst = out + ((size_t)b * T_ + l) * D_;
    for (int k = tid * 4; k < D_; k += 1024)
        *(float4*)(dst + k) = *(const float4*)(src + k);
}

// ---------------------------------------------------------------------------
// Kernel 4: token embedding gather for the non-audio suffix t in [L, T)
// (modality_mask is structurally arange(T) < L)
// ---------------------------------------------------------------------------
__global__ void __launch_bounds__(256) text_embed(
    const float* __restrict__ w, const int* __restrict__ ids,
    float* __restrict__ out)
{
    const int blk = blockIdx.x;                 // 0 .. B*(T-L)-1
    const int b = blk / (T_ - L_);
    const int t = L_ + blk % (T_ - L_);
    const int tok = ids[b * T_ + t];
    const float* src = w + (size_t)tok * D_;
    float* dst = out + ((size_t)b * T_ + t) * D_;
    for (int k = threadIdx.x * 4; k < D_; k += 1024)
        *(float4*)(dst + k) = *(const float4*)(src + k);
}

// ---------------------------------------------------------------------------
extern "C" void kernel_launch(void* const* d_in, const int* in_sizes, int n_in,
                              void* d_out, int out_size, void* d_ws, size_t ws_size,
                              hipStream_t stream) {
    const float* enc = (const float*)d_in[0];     // (B,L,D) f32
    const float* w   = (const float*)d_in[1];     // (V,D)   f32
    const int*   ids = (const int*)d_in[2];       // (B,T)   int
    (void)in_sizes; (void)n_in; (void)ws_size; (void)out_size;

    unsigned short* a_bf16 = (unsigned short*)d_ws;                   // 8.4 MB
    float2* partial = (float2*)((char*)d_ws + (size_t)M_ * D_ * 2);   // 32 MB

    cvt_a_bf16<<<(M_ * D_) / 1024, 256, 0, stream>>>(enc, a_bf16);

    const size_t lds = (size_t)COLS_PER_WG * D_ * 2 + COLS_PER_WG * sizeof(float);
    sim_argmax<<<NCHUNK, 256, lds, stream>>>(a_bf16, w, partial);

    reduce_quant<<<M_, 256, 0, stream>>>(partial, w, (float*)d_out);
    text_embed<<<B_ * (T_ - L_), 256, 0, stream>>>(w, ids, (float*)d_out);
}
